// CardEncoder_627065225500
// MI455X (gfx1250) — matrix operand order repaired and checked by reading, not compiled, not run on hardware
//
#include <hip/hip_runtime.h>

#define G  4
#define NN 50000
#define EE 400000
#define D  128
#define L  3
#define NEXP 8
#define AH 64
#define LN_EPS 1e-5f
#define PAD 8

#define MT    80            // nodes per block (5 x 16 m-subtiles)
#define MSUB  (MT / 16)     // 5

typedef __attribute__((ext_vector_type(16))) _Float16 v16h;
typedef __attribute__((ext_vector_type(8)))  float    v8f;

__device__ __forceinline__ void atomAddF32(float* p, float v) {
  __hip_atomic_fetch_add(p, v, __ATOMIC_RELAXED, __HIP_MEMORY_SCOPE_AGENT);
}

// -------- weight f32[K][N] -> f16 transposed WT[N][K] (once per call) --------
__global__ void wconv_kernel(const float* __restrict__ W1, const float* __restrict__ W2,
                             _Float16* __restrict__ W1t, _Float16* __restrict__ W2t, int n) {
  const int i = blockIdx.x * 256 + threadIdx.x;
  if (i < n) {
    const int l   = i / (D * D);
    const int rem = i - l * D * D;
    const int k   = rem >> 7;       // rem / D
    const int c   = rem & (D - 1);  // rem % D
    const int o   = l * D * D + c * D + k;
    W1t[o] = (_Float16)W1[i];
    W2t[o] = (_Float16)W2[i];
  }
}

// ---------------- edge scatter-add: agg[dst] += x[src] ----------------
// one wave per edge; lane owns a contiguous float4 of the 512B feature row
__global__ void scatter_agg_kernel(const float* __restrict__ x, const int* __restrict__ ei,
                                   float* __restrict__ agg) {
  const int g    = blockIdx.y;
  const int wave = threadIdx.x >> 5;
  const int lane = threadIdx.x & 31;
  const int e    = blockIdx.x * 8 + wave;          // grid.x = EE/8
  const int* eig = ei + (size_t)g * 2 * EE;
  const int src = eig[e];
  const int dst = eig[EE + e];
  const float4 v = *(const float4*)(x + ((size_t)g * NN + src) * D + lane * 4);
  float* a = agg + ((size_t)g * NN + dst) * D + lane * 4;
  atomAddF32(a + 0, v.x);
  atomAddF32(a + 1, v.y);
  atomAddF32(a + 2, v.z);
  atomAddF32(a + 3, v.w);
}

// ------- fused GIN MLP: (x+agg)@W1+b1, relu, @W2+b2, (+x), LayerNorm -------
// block = 80 nodes, 256 threads = 8 waves; wave w owns output cols [16w,16w+16)
// and iterates 5 m-subtiles, reusing each B fragment 5x.
__global__ void __launch_bounds__(256, 2) gin_mlp_kernel(
    const float* __restrict__ xin, const float* __restrict__ agg,
    const _Float16* __restrict__ W1t, const float* __restrict__ b1,
    const _Float16* __restrict__ W2t, const float* __restrict__ b2,
    const float* __restrict__ lng, const float* __restrict__ lnb,
    float* __restrict__ xout, int do_res)
{
  __shared__ _Float16 shA[MT][D + PAD];   // layer input tile (f16)
  __shared__ _Float16 shM[MT][D + PAD];   // relu(h@W1+b1) (f16)
  __shared__ float    shO[MT][D + PAD];   // final pre-LN (f32)
  __shared__ float    shMu[MT], shRs[MT];

  const int g     = blockIdx.y;
  const int node0 = blockIdx.x * MT;
  const int tid   = threadIdx.x;
  const size_t base = ((size_t)g * NN + node0) * D;

  // stage tile: h = x + agg -> f16   (MT*16 = 1280 chunks of 8 floats, 5/thread)
#pragma unroll
  for (int it = 0; it < MT * 16 / 256; ++it) {
    const int idx = it * 256 + tid;
    const int row = idx >> 4;
    const int c0  = (idx & 15) * 8;
    const float* px = xin + base + (size_t)row * D + c0;
    const float* pa = agg + base + (size_t)row * D + c0;
#pragma unroll
    for (int j = 0; j < 8; ++j)
      shA[row][c0 + j] = (_Float16)(px[j] + pa[j]);
  }
  __syncthreads();

  const int wave  = tid >> 5;
  const int lane  = tid & 31;
  const int hi    = lane >> 4;
  const int lrow  = lane & 15;
  const int nbase = wave * 16;
  const int col   = nbase + lrow;

  // ---- GEMM1: shA[MT x K] @ W1 (cols nbase..nbase+15), via WT[N][K] ----
  v8f acc[MSUB];
#pragma unroll
  for (int mt = 0; mt < MSUB; ++mt) acc[mt] = (v8f){};

  const _Float16* w1row = W1t + (size_t)col * D;   // contiguous K for this lane's N
#pragma unroll
  for (int ks = 0; ks < D / 32; ++ks) {
    const int kb = ks * 32;
    // B fragment: element i -> K = kb + i + 16*hi  => 16 contiguous halfs (32B aligned)
    const v16h b = *(const v16h*)(w1row + kb + hi * 16);
#pragma unroll
    for (int mt = 0; mt < MSUB; ++mt) {
      const int arow = mt * 16 + lrow;
      v16h a;
#pragma unroll
      for (int j = 0; j < 8; ++j) {
        a[j]     = shA[arow][kb + hi * 16 + j];       // K = kb+16*hi+j
        a[j + 8] = shA[arow][kb + hi * 16 + 8 + j];   // K = kb+16*hi+8+j
      }
      acc[mt] = __builtin_amdgcn_wmma_f32_16x16x32_f16(false, a, false, b,
                                                       (short)0, acc[mt], false, false);
    }
  }
  // bias + relu -> shM  (C/D layout: elem r -> M = r + 8*hi, N = lane&15)
  {
    const float bb = b1[col];
#pragma unroll
    for (int mt = 0; mt < MSUB; ++mt)
#pragma unroll
      for (int r = 0; r < 8; ++r) {
        float v = acc[mt][r] + bb;
        v = v > 0.f ? v : 0.f;
        shM[mt * 16 + r + hi * 8][col] = (_Float16)v;
      }
  }
  __syncthreads();

  // ---- GEMM2 ----
#pragma unroll
  for (int mt = 0; mt < MSUB; ++mt) acc[mt] = (v8f){};
  const _Float16* w2row = W2t + (size_t)col * D;
#pragma unroll
  for (int ks = 0; ks < D / 32; ++ks) {
    const int kb = ks * 32;
    const v16h b = *(const v16h*)(w2row + kb + hi * 16);
#pragma unroll
    for (int mt = 0; mt < MSUB; ++mt) {
      const int arow = mt * 16 + lrow;
      v16h a;
#pragma unroll
      for (int j = 0; j < 8; ++j) {
        a[j]     = shM[arow][kb + hi * 16 + j];
        a[j + 8] = shM[arow][kb + hi * 16 + 8 + j];
      }
      acc[mt] = __builtin_amdgcn_wmma_f32_16x16x32_f16(false, a, false, b,
                                                       (short)0, acc[mt], false, false);
    }
  }
  // bias + optional residual -> shO
  {
    const float bb = b2[col];
#pragma unroll
    for (int mt = 0; mt < MSUB; ++mt)
#pragma unroll
      for (int r = 0; r < 8; ++r) {
        const int mrow = mt * 16 + r + hi * 8;
        float v = acc[mt][r] + bb;
        if (do_res) v += xin[base + (size_t)mrow * D + col];
        shO[mrow][col] = v;
      }
  }
  __syncthreads();

  // ---- LayerNorm over D per node row ----
  if (tid < MT) {
    float s = 0.f, ss = 0.f;
#pragma unroll 4
    for (int c = 0; c < D; ++c) { const float v = shO[tid][c]; s += v; ss += v * v; }
    const float mu  = s * (1.0f / D);
    const float var = ss * (1.0f / D) - mu * mu;
    shMu[tid] = mu;
    shRs[tid] = rsqrtf(var + LN_EPS);
  }
  __syncthreads();
#pragma unroll
  for (int it = 0; it < MT * 16 / 256; ++it) {
    const int idx = it * 256 + tid;
    const int row = idx >> 4;
    const int c0  = (idx & 15) * 8;
    const float mu = shMu[row], rs = shRs[row];
    float* po = xout + base + (size_t)row * D + c0;
#pragma unroll
    for (int j = 0; j < 8; ++j) {
      const int c = c0 + j;
      po[j] = (shO[row][c] - mu) * rs * lng[c] + lnb[c];
    }
  }
}

// ---------------- readout: gvec[g][d] = sum_n x[g][n][d] ----------------
#define RD_CHUNK 500
__global__ void readout_kernel(const float* __restrict__ x, float* __restrict__ gvec) {
  const int g = blockIdx.y;
  const int d = threadIdx.x;                 // 128 threads
  const int n0 = blockIdx.x * RD_CHUNK;      // grid.x = NN/RD_CHUNK
  float s = 0.f;
  const float* p = x + ((size_t)g * NN + n0) * D + d;
  for (int n = 0; n < RD_CHUNK; ++n, p += D) s += *p;
  atomAddF32(&gvec[g * D + d], s);
}

// ---------------- attention pooling (tiny, one block) ----------------
__global__ void attention_kernel(const float* __restrict__ gvec,
                                 const float* __restrict__ w1, const float* __restrict__ w2,
                                 float* __restrict__ out) {
  __shared__ float sup[AH][G];
  __shared__ float sc[NEXP][G];
  __shared__ float att[NEXP][G];
  const int t = threadIdx.x;
  {
    const int ah = t >> 2, gg = t & 3;      // 64*4 = 256 entries
    float s = 0.f;
    for (int d2 = 0; d2 < D; ++d2) s += w1[ah * D + d2] * gvec[gg * D + d2];
    sup[ah][gg] = tanhf(s);
  }
  __syncthreads();
  if (t < NEXP * G) {
    const int ne = t >> 2, gg = t & 3;
    float s = 0.f;
    for (int a2 = 0; a2 < AH; ++a2) s += w2[ne * AH + a2] * sup[a2][gg];
    sc[ne][gg] = s;
  }
  __syncthreads();
  if (t < NEXP * G) {
    const int ne = t >> 2, gg = t & 3;
    float m = sc[ne][0];
    for (int k = 1; k < G; ++k) m = fmaxf(m, sc[ne][k]);
    float den = 0.f;
    for (int k = 0; k < G; ++k) den += __expf(sc[ne][k] - m);
    att[ne][gg] = __expf(sc[ne][gg] - m) / den;
  }
  __syncthreads();
#pragma unroll
  for (int j = 0; j < 4; ++j) {
    const int idx = t * 4 + j;               // 1024 outputs
    const int ne = idx >> 7, d2 = idx & 127;
    float s = 0.f;
    for (int gg = 0; gg < G; ++gg) s += att[ne][gg] * gvec[gg * D + d2];
    out[idx] = s;
  }
}

extern "C" void kernel_launch(void* const* d_in, const int* in_sizes, int n_in,
                              void* d_out, int out_size, void* d_ws, size_t ws_size,
                              hipStream_t stream) {
  const float* x0  = (const float*)d_in[0];
  const int*   ei  = (const int*)d_in[1];
  // d_in[2] = edge_attr: unused by the conv path
  const float* Ws1 = (const float*)d_in[3];
  const float* bs1 = (const float*)d_in[4];
  const float* Ws2 = (const float*)d_in[5];
  const float* bs2 = (const float*)d_in[6];
  const float* lng = (const float*)d_in[7];
  const float* lnb = (const float*)d_in[8];
  const float* aw1 = (const float*)d_in[9];
  const float* aw2 = (const float*)d_in[10];

  char* ws = (char*)d_ws;
  const size_t xbytes = (size_t)G * NN * D * sizeof(float);
  float*    xA   = (float*)ws;                ws += xbytes;
  float*    xB   = (float*)ws;                ws += xbytes;
  float*    agg  = (float*)ws;                ws += xbytes;
  _Float16* W1t  = (_Float16*)ws;             ws += (size_t)L * D * D * sizeof(_Float16);
  _Float16* W2t  = (_Float16*)ws;             ws += (size_t)L * D * D * sizeof(_Float16);
  float*    gvec = (float*)ws;                ws += (size_t)G * D * sizeof(float);

  const int wn = L * D * D;
  wconv_kernel<<<(wn + 255) / 256, 256, 0, stream>>>(Ws1, Ws2, W1t, W2t, wn);

  float* bufs[2] = { xA, xB };
  const float* cur = x0;
  for (int l = 0; l < L; ++l) {
    float* nxt = bufs[l & 1];
    hipMemsetAsync(agg, 0, xbytes, stream);
    scatter_agg_kernel<<<dim3(EE / 8, G), 256, 0, stream>>>(cur, ei, agg);
    gin_mlp_kernel<<<dim3(NN / MT, G), 256, 0, stream>>>(
        cur, agg,
        W1t + (size_t)l * D * D, bs1 + l * D,
        W2t + (size_t)l * D * D, bs2 + l * D,
        lng + l * D, lnb + l * D,
        nxt, (l < L - 1) ? 1 : 0);
    cur = nxt;
  }

  hipMemsetAsync(gvec, 0, (size_t)G * D * sizeof(float), stream);
  readout_kernel<<<dim3(NN / RD_CHUNK, G), D, 0, stream>>>(cur, gvec);
  attention_kernel<<<1, 256, 0, stream>>>(gvec, aw1, aw2, (float*)d_out);
}
